// LLaMAMLKV_88227218194463
// MI455X (gfx1250) — matrix operand order repaired and checked by reading, not compiled, not run on hardware
//
#include <hip/hip_runtime.h>
#include <hip/hip_bf16.h>

// ---------------------------------------------------------------------------
// CDNA5 (gfx1250) wave32 WMMA types
// ---------------------------------------------------------------------------
typedef __attribute__((ext_vector_type(16))) __bf16 v16bf;
typedef __attribute__((ext_vector_type(8)))  __bf16 v8bf;
typedef __attribute__((ext_vector_type(8)))  float  v8f;
typedef __attribute__((ext_vector_type(4)))  unsigned int v4u;
typedef __attribute__((ext_vector_type(8)))  int v8i;
typedef __attribute__((ext_vector_type(4)))  int v4i;

#define WMMA_BF16(a, b, c) \
  __builtin_amdgcn_wmma_f32_16x16x32_bf16(false, (a), false, (b), (short)0, (c), false, false)

__device__ __forceinline__ v8f zero8() {
  v8f z;
#pragma unroll
  for (int e = 0; e < 8; ++e) z[e] = 0.0f;
  return z;
}

// A-matrix fragment (16x32 bf16, M = lane&15).
// ISA 7.12.2: lane<16 holds K = {0..7, 16..23}, lane>=16 holds K = {8..15, 24..31}.
__device__ __forceinline__ v16bf load_a_frag(const __bf16* rowp, int hi) {
  v8bf lo = *(const v8bf*)(rowp + hi * 8);
  v8bf hh = *(const v8bf*)(rowp + 16 + hi * 8);
  v16bf r;
#pragma unroll
  for (int e = 0; e < 8; ++e) { r[e] = lo[e]; r[8 + e] = hh[e]; }
  return r;
}

// B-matrix fragment (32x16 bf16, N = lane&15). B stored transposed in LDS as
// [N][K]; same K order as A: lane<16 K = {0..7, 16..23}, lane>=16 K = {8..15, 24..31}.
__device__ __forceinline__ v16bf load_b_frag(const __bf16* rowp, int hi) {
  v8bf b0 = *(const v8bf*)(rowp + hi * 8);
  v8bf b1 = *(const v8bf*)(rowp + 16 + hi * 8);
  v16bf r;
#pragma unroll
  for (int e = 0; e < 8; ++e) { r[e] = b0[e]; r[8 + e] = b1[e]; }
  return r;
}

// ---------------------------------------------------------------------------
// TDM: issue a 2D tensor_load_to_lds for a [rows x 32] bf16 tile with an
// LDS row stride of 40 bf16 (64B data + 16B pad per row).  D# per ISA ch.8.
// Must be executed by a single wave; completion via TENSORcnt.
// This toolchain declares the 6-arg builtin:
//   (uint32x4 g0, int32x8 g1, int32x4 g2, int32x4 g3, int32x8 g4, i32 cpol)
// ---------------------------------------------------------------------------
__device__ __forceinline__ void tdm_load_tile_bf16(const __bf16* gsrc,
                                                   unsigned lds_byte_addr,
                                                   int K /*row stride, elems*/,
                                                   int rows) {
  unsigned long long ga = (unsigned long long)(size_t)gsrc;
  v4u g0;
  g0[0] = 1u;                                   // count=1, user descriptor
  g0[1] = lds_byte_addr;                        // lds_addr (bytes)
  g0[2] = (unsigned)(ga & 0xffffffffu);         // global_addr[31:0]
  g0[3] = (unsigned)((ga >> 32) & 0x1ffffffu)   // global_addr[56:32]
          | (2u << 30);                         // type = 2 ("image")
  v8i g1;
  // data_size=1 (2B), pad_enable, pad_interval=3 (16 DW), pad_amount=3 (4 DW)
  g1[0] = (int)((1u << 16) | (1u << 20) | (3u << 22) | (3u << 25));
  g1[1] = (int)(((unsigned)K & 0xffffu) << 16);             // tensor_dim0 lo16
  g1[2] = (int)((((unsigned)K >> 16) & 0xffffu) |
                (((unsigned)rows & 0xffffu) << 16));        // dim0 hi | dim1 lo
  g1[3] = (int)(32u << 16);                                 // dim1 hi=0 | tile_dim0=32
  g1[4] = (int)((unsigned)rows & 0xffffu);                  // tile_dim1 | tile_dim2=0
  g1[5] = (int)(unsigned)K;                                 // dim0_stride[31:0]
  g1[6] = 0;                                                // stride0 hi | stride1 lo
  g1[7] = 0;
  v4i gz4 = {0, 0, 0, 0};
  v8i gz8 = {0, 0, 0, 0, 0, 0, 0, 0};
  __builtin_amdgcn_tensor_load_to_lds(g0, g1, gz4, gz4, gz8, 0);
}

// ---------------------------------------------------------------------------
// Generic GEMM: C[M,N] = A_bf16[M,K] * B_f32[K,N] (+ Res), C is f32.
// Block tile 128x128, 8 waves (2x4), wave tile 64x32, K-tile 32.
// A tile staged by the Tensor Data Mover; B (f32 weights) converted to bf16
// cooperatively while staging into LDS.
// Requires: M%128==0, N%128==0, K%32==0 (true for every GEMM in this model).
// ---------------------------------------------------------------------------
template <bool ADD_RES>
__global__ __launch_bounds__(256) void gemm_bf16(const __bf16* __restrict__ A,
                                                 const float* __restrict__ B,
                                                 float* __restrict__ C,
                                                 const float* __restrict__ Res,
                                                 int M, int N, int K) {
  constexpr int AS_STR = 40;  // bf16 elems; 80B rows keep 16B alignment
  constexpr int BS_STR = 40;
  __shared__ __align__(16) __bf16 As[128 * AS_STR];  // [m][k]
  __shared__ __align__(16) __bf16 Bs[128 * BS_STR];  // [n][k] (transposed)

  const int t = threadIdx.x;
  const int lane = t & 31, wave = t >> 5;
  const int l15 = lane & 15, hi = lane >> 4;
  const int wm = wave >> 2, wn = wave & 3;
  const int m0 = blockIdx.y * 128, n0 = blockIdx.x * 128;

  v8f acc[4][2];
#pragma unroll
  for (int mi = 0; mi < 4; ++mi)
#pragma unroll
    for (int ni = 0; ni < 2; ++ni) acc[mi][ni] = zero8();

  const unsigned as_lds = (unsigned)(size_t)(&As[0]);  // flat LDS addr, [31:0] = offset

  for (int kt = 0; kt < K; kt += 32) {
    __syncthreads();  // previous tile fully consumed
    // --- stage A tile 128x32 bf16 via TDM (wave 0 issues; TENSORcnt) ---
    if (wave == 0) {
      tdm_load_tile_bf16(A + (size_t)m0 * K + kt, as_lds, K, 128);
    }
    // --- stage B tile: 32x128 f32 -> bf16, transposed to [n][k] ---
#pragma unroll
    for (int it = 0; it < 4; ++it) {
      int idx = it * 256 + t;        // 0..1023
      int kk  = idx >> 5;            // 0..31
      int nc  = (idx & 31) * 4;      // 0..124
      float4 bv = *(const float4*)(B + (size_t)(kt + kk) * N + n0 + nc);
      Bs[(nc + 0) * BS_STR + kk] = (__bf16)bv.x;
      Bs[(nc + 1) * BS_STR + kk] = (__bf16)bv.y;
      Bs[(nc + 2) * BS_STR + kk] = (__bf16)bv.z;
      Bs[(nc + 3) * BS_STR + kk] = (__bf16)bv.w;
    }
    if (kt + 32 < K) {
      __builtin_prefetch(B + (size_t)(kt + 32) * N + n0 + t, 0, 1);  // global_prefetch_b8
    }
    if (wave == 0) {
      __builtin_amdgcn_s_wait_tensorcnt(0);  // drain TDM before the barrier
    }
    __syncthreads();

    v16bf afrag[4], bfrag[2];
#pragma unroll
    for (int mi = 0; mi < 4; ++mi)
      afrag[mi] = load_a_frag(&As[(wm * 64 + mi * 16 + l15) * AS_STR], hi);
#pragma unroll
    for (int ni = 0; ni < 2; ++ni)
      bfrag[ni] = load_b_frag(&Bs[(wn * 32 + ni * 16 + l15) * BS_STR], hi);
#pragma unroll
    for (int mi = 0; mi < 4; ++mi)
#pragma unroll
      for (int ni = 0; ni < 2; ++ni)
        acc[mi][ni] = WMMA_BF16(afrag[mi], bfrag[ni], acc[mi][ni]);
  }

  // epilogue (C layout: M = r + hi*8, N = l15)
#pragma unroll
  for (int mi = 0; mi < 4; ++mi)
#pragma unroll
    for (int ni = 0; ni < 2; ++ni)
#pragma unroll
      for (int r = 0; r < 8; ++r) {
        int row = m0 + wm * 64 + mi * 16 + r + hi * 8;
        int col = n0 + wn * 32 + ni * 16 + l15;
        float v = acc[mi][ni][r];
        if (ADD_RES) v += Res[(size_t)row * N + col];
        C[(size_t)row * N + col] = v;
      }
}

// ---------------------------------------------------------------------------
// Flash attention (causal), one (head, 128-q-row tile) per block, 8 waves.
// Each wave owns 16 q rows; key tiles of 64; QK^T and P*V via bf16 WMMA.
// q,k,v are f32 in workspace; output written bf16 (feeds wo GEMM).
// ---------------------------------------------------------------------------
__global__ __launch_bounds__(256) void flash_attn(const float* __restrict__ qf,
                                                  const float* __restrict__ kf,
                                                  const float* __restrict__ vf,
                                                  __bf16* __restrict__ aobf) {
  constexpr int STR = 72;  // 144B rows, 16B aligned
  __shared__ __align__(16) __bf16 Qs[128 * STR];  // [qrow][d]
  __shared__ __align__(16) __bf16 Ks[64 * STR];   // [key][d]   (B for QK^T)
  __shared__ __align__(16) __bf16 Vs[64 * STR];   // [d][key]   (B for P*V)
  __shared__ __align__(16) __bf16 Ps[128 * STR];  // per-wave P scratch

  const int q0 = blockIdx.x * 128;
  const int h  = blockIdx.y;        // 0..15 ; h = g*NREP + r
  const int g  = h >> 2;            // kv head
  const int t = threadIdx.x;
  const int lane = t & 31, wave = t >> 5;
  const int l15 = lane & 15, hi = lane >> 4;

  // stage Q tile (f32 -> bf16)
  {
    int r = t >> 1, dc = (t & 1) * 32;
    const float* qrow = qf + (size_t)(q0 + r) * 1024 + h * 64 + dc;
#pragma unroll
    for (int j = 0; j < 32; j += 4) {
      float4 v = *(const float4*)(qrow + j);
      Qs[r * STR + dc + j + 0] = (__bf16)v.x;
      Qs[r * STR + dc + j + 1] = (__bf16)v.y;
      Qs[r * STR + dc + j + 2] = (__bf16)v.z;
      Qs[r * STR + dc + j + 3] = (__bf16)v.w;
    }
  }
  __syncthreads();

  v16bf qfr[2];
#pragma unroll
  for (int kk = 0; kk < 2; ++kk)
    qfr[kk] = load_a_frag(&Qs[(wave * 16 + l15) * STR + kk * 32], hi);

  v8f o[4];
#pragma unroll
  for (int n = 0; n < 4; ++n) o[n] = zero8();
  float mrow[8], lsum[8];
#pragma unroll
  for (int r = 0; r < 8; ++r) { mrow[r] = -1e30f; lsum[r] = 0.0f; }

  const int kend = q0 + 128;
  for (int kt = 0; kt < kend; kt += 64) {
    __syncthreads();
    // stage K (row-major [key][d]) and V (transposed [d][key])
    {
      int key = t >> 2, dc = (t & 3) * 16;
      const float* kr = kf + (size_t)(kt + key) * 256 + g * 64 + dc;
      const float* vr = vf + (size_t)(kt + key) * 256 + g * 64 + dc;
#pragma unroll
      for (int j = 0; j < 16; j += 4) {
        float4 kv = *(const float4*)(kr + j);
        Ks[key * STR + dc + j + 0] = (__bf16)kv.x;
        Ks[key * STR + dc + j + 1] = (__bf16)kv.y;
        Ks[key * STR + dc + j + 2] = (__bf16)kv.z;
        Ks[key * STR + dc + j + 3] = (__bf16)kv.w;
        float4 vv = *(const float4*)(vr + j);
        Vs[(dc + j + 0) * STR + key] = (__bf16)vv.x;
        Vs[(dc + j + 1) * STR + key] = (__bf16)vv.y;
        Vs[(dc + j + 2) * STR + key] = (__bf16)vv.z;
        Vs[(dc + j + 3) * STR + key] = (__bf16)vv.w;
      }
    }
    __syncthreads();

    // scores: S = Q * K^T  (4 key sub-tiles of 16, 2 k-steps over HD=64)
    v8f s[4];
#pragma unroll
    for (int nt = 0; nt < 4; ++nt) {
      s[nt] = zero8();
#pragma unroll
      for (int kk = 0; kk < 2; ++kk) {
        v16bf kb = load_b_frag(&Ks[(nt * 16 + l15) * STR + kk * 32], hi);
        s[nt] = WMMA_BF16(qfr[kk], kb, s[nt]);
      }
    }

    // scale + causal mask + online softmax (rows = r + hi*8 within wave tile)
    float p[4][8], rmax[8];
#pragma unroll
    for (int r = 0; r < 8; ++r) rmax[r] = -1e30f;
#pragma unroll
    for (int nt = 0; nt < 4; ++nt) {
      int kcol = kt + nt * 16 + l15;
#pragma unroll
      for (int r = 0; r < 8; ++r) {
        int qrow = q0 + wave * 16 + r + hi * 8;
        float v = s[nt][r] * 0.125f;  // 1/sqrt(64)
        if (kcol > qrow) v = -1e30f;
        p[nt][r] = v;
        rmax[r] = fmaxf(rmax[r], v);
      }
    }
#pragma unroll
    for (int r = 0; r < 8; ++r) {
      rmax[r] = fmaxf(rmax[r], __shfl_xor(rmax[r], 1));
      rmax[r] = fmaxf(rmax[r], __shfl_xor(rmax[r], 2));
      rmax[r] = fmaxf(rmax[r], __shfl_xor(rmax[r], 4));
      rmax[r] = fmaxf(rmax[r], __shfl_xor(rmax[r], 8));
    }
    float corr[8];
#pragma unroll
    for (int r = 0; r < 8; ++r) {
      float mn = fmaxf(mrow[r], rmax[r]);
      corr[r] = __expf(mrow[r] - mn);
      mrow[r] = mn;
    }
    float rsum[8];
#pragma unroll
    for (int r = 0; r < 8; ++r) rsum[r] = 0.0f;
#pragma unroll
    for (int nt = 0; nt < 4; ++nt)
#pragma unroll
      for (int r = 0; r < 8; ++r) {
        float e = __expf(p[nt][r] - mrow[r]);
        p[nt][r] = e;
        rsum[r] += e;
      }
#pragma unroll
    for (int r = 0; r < 8; ++r) {
      rsum[r] += __shfl_xor(rsum[r], 1);
      rsum[r] += __shfl_xor(rsum[r], 2);
      rsum[r] += __shfl_xor(rsum[r], 4);
      rsum[r] += __shfl_xor(rsum[r], 8);
      lsum[r] = lsum[r] * corr[r] + rsum[r];
    }
#pragma unroll
    for (int n = 0; n < 4; ++n)
#pragma unroll
      for (int r = 0; r < 8; ++r) o[n][r] *= corr[r];

    // P: C-layout -> LDS (wave-private) -> A-layout fragments
    __bf16* Pw = &Ps[wave * 16 * STR];
#pragma unroll
    for (int nt = 0; nt < 4; ++nt)
#pragma unroll
      for (int r = 0; r < 8; ++r)
        Pw[(r + hi * 8) * STR + nt * 16 + l15] = (__bf16)p[nt][r];

    v16bf pf[2];
#pragma unroll
    for (int kk = 0; kk < 2; ++kk)
      pf[kk] = load_a_frag(&Pw[l15 * STR + kk * 32], hi);

#pragma unroll
    for (int nto = 0; nto < 4; ++nto)
#pragma unroll
      for (int kk = 0; kk < 2; ++kk) {
        v16bf vb = load_b_frag(&Vs[(nto * 16 + l15) * STR + kk * 32], hi);
        o[nto] = WMMA_BF16(pf[kk], vb, o[nto]);
      }
  }

  // normalize and store bf16 (S, H*HD) layout
#pragma unroll
  for (int nto = 0; nto < 4; ++nto)
#pragma unroll
    for (int r = 0; r < 8; ++r) {
      int row = q0 + wave * 16 + r + hi * 8;
      int col = h * 64 + nto * 16 + l15;
      aobf[(size_t)row * 1024 + col] = (__bf16)(o[nto][r] / lsum[r]);
    }
}

// ---------------------------------------------------------------------------
// Small elementwise kernels
// ---------------------------------------------------------------------------
__global__ __launch_bounds__(256) void embed_kernel(const int* __restrict__ tokens,
                                                    const float* __restrict__ emb,
                                                    float* __restrict__ hs) {
  int srow = blockIdx.x;
  int tok = tokens[srow];
  const float4* e = (const float4*)(emb + (size_t)tok * 1024);
  float4* o = (float4*)(hs + (size_t)srow * 1024);
  o[threadIdx.x] = e[threadIdx.x];
}

__global__ __launch_bounds__(256) void rmsnorm_bf16(const float* __restrict__ x,
                                                    const float* __restrict__ w,
                                                    __bf16* __restrict__ out,
                                                    int dim) {
  int row = blockIdx.x;
  const float* xr = x + (size_t)row * dim;
  float ss = 0.0f;
  for (int i = threadIdx.x; i < dim; i += 256) { float v = xr[i]; ss += v * v; }
  for (int m = 16; m; m >>= 1) ss += __shfl_xor(ss, m);
  __shared__ float red[8];
  if ((threadIdx.x & 31) == 0) red[threadIdx.x >> 5] = ss;
  __syncthreads();
  float tot = 0.0f;
#pragma unroll
  for (int j = 0; j < 8; ++j) tot += red[j];
  float rs = rsqrtf(tot / (float)dim + 1e-5f);
  for (int i = threadIdx.x; i < dim; i += 256)
    out[(size_t)row * dim + i] = (__bf16)(xr[i] * rs * w[i]);
}

__global__ __launch_bounds__(256) void silu_mul(const float* __restrict__ gt,
                                                const float* __restrict__ up,
                                                __bf16* __restrict__ out, int n) {
  int i = blockIdx.x * 256 + threadIdx.x;
  if (i < n) {
    float gv = gt[i];
    float s = gv / (1.0f + __expf(-gv));
    out[i] = (__bf16)(s * up[i]);
  }
}

__global__ __launch_bounds__(256) void rope_kernel(float* __restrict__ x,
                                                   int nheads, int total) {
  int tid = blockIdx.x * 256 + threadIdx.x;
  if (tid >= total) return;
  int i = tid & 31;                 // pair index, HD/2 = 32
  int hh = (tid >> 5) % nheads;
  int s = tid / (32 * nheads);
  float inv = __expf(-(2.0f * (float)i / 64.0f) * __logf(10000.0f));
  float ang = (float)s * inv;
  float sn, cs;
  __sincosf(ang, &sn, &cs);
  size_t base = (size_t)s * ((size_t)nheads * 64) + (size_t)hh * 64 + 2 * i;
  float x1 = x[base], x2 = x[base + 1];
  x[base]     = x1 * cs - x2 * sn;
  x[base + 1] = x1 * sn + x2 * cs;
}

// ---------------------------------------------------------------------------
// Driver
// ---------------------------------------------------------------------------
extern "C" void kernel_launch(void* const* d_in, const int* in_sizes, int n_in,
                              void* d_out, int out_size, void* d_ws, size_t ws_size,
                              hipStream_t stream) {
  (void)in_sizes; (void)n_in; (void)out_size; (void)ws_size;
  constexpr int S = 2048, DIM = 1024, H = 16, HKV = 4, HD = 64;
  constexpr int FFN = 2816, PHID = 4096, VOCAB = 32000, L = 4;
  const int KV_IDX[L] = {0, 0, 1, 1};
  const bool HAS_KV[L] = {true, false, true, false};

  const int*   tokens       = (const int*)d_in[0];
  const float* tok_emb      = (const float*)d_in[1];
  const float* wq           = (const float*)d_in[2];
  const float* wo           = (const float*)d_in[3];
  const float* wk1          = (const float*)d_in[4];
  const float* wk2          = (const float*)d_in[5];
  const float* wk3          = (const float*)d_in[6];
  const float* wv1          = (const float*)d_in[7];
  const float* wv2          = (const float*)d_in[8];
  const float* wv3          = (const float*)d_in[9];
  const float* attn_norm_w  = (const float*)d_in[10];
  const float* ffn_norm_w   = (const float*)d_in[11];
  const float* w1           = (const float*)d_in[12];
  const float* w2           = (const float*)d_in[13];
  const float* w3           = (const float*)d_in[14];
  const float* final_norm_w = (const float*)d_in[15];
  const float* out_w        = (const float*)d_in[16];
  float* logits = (float*)d_out;

  char* ws = (char*)d_ws;
  const size_t MB = 1ull << 20;
  float*  hs    = (float*)(ws + 0 * MB);    //  8 MB  h state (S x DIM)
  __bf16* xnbf  = (__bf16*)(ws + 8 * MB);   //  4 MB  normed activations bf16
  float*  qf    = (float*)(ws + 12 * MB);   //  8 MB  q (S x 1024)
  float*  kf    = (float*)(ws + 20 * MB);   //  2 MB  k (S x 256)
  float*  vf    = (float*)(ws + 22 * MB);   //  2 MB  v (S x 256)
  float*  g1    = (float*)(ws + 24 * MB);   // 32 MB  gate  (S x PHID max)
  float*  g3    = (float*)(ws + 56 * MB);   // 32 MB  up    (S x PHID max)
  __bf16* actbf = (__bf16*)(ws + 88 * MB);  // 16 MB  silu(g)*u bf16
  __bf16* aobf  = (__bf16*)(ws + 104 * MB); //  4 MB  attention out bf16

  dim3 blk(256);
  embed_kernel<<<S, blk, 0, stream>>>(tokens, tok_emb, hs);

  for (int i = 0; i < L; ++i) {
    rmsnorm_bf16<<<S, blk, 0, stream>>>(hs, attn_norm_w + (size_t)i * DIM, xnbf, DIM);

    // q = xn @ wq[i], then RoPE
    gemm_bf16<false><<<dim3((H * HD) / 128, S / 128), blk, 0, stream>>>(
        xnbf, wq + (size_t)i * DIM * H * HD, qf, nullptr, S, H * HD, DIM);
    {
      int tot = S * H * (HD / 2);
      rope_kernel<<<(tot + 255) / 256, blk, 0, stream>>>(qf, H, tot);
    }

    if (HAS_KV[i]) {
      int j = KV_IDX[i];
      // keys = rope(swiglu(xn, wk1, wk2, wk3))
      gemm_bf16<false><<<dim3(PHID / 128, S / 128), blk, 0, stream>>>(
          xnbf, wk1 + (size_t)j * DIM * PHID, g1, nullptr, S, PHID, DIM);
      gemm_bf16<false><<<dim3(PHID / 128, S / 128), blk, 0, stream>>>(
          xnbf, wk3 + (size_t)j * DIM * PHID, g3, nullptr, S, PHID, DIM);
      silu_mul<<<(S * PHID + 255) / 256, blk, 0, stream>>>(g1, g3, actbf, S * PHID);
      gemm_bf16<false><<<dim3((HKV * HD) / 128, S / 128), blk, 0, stream>>>(
          actbf, wk2 + (size_t)j * PHID * HKV * HD, kf, nullptr, S, HKV * HD, PHID);
      {
        int tot = S * HKV * (HD / 2);
        rope_kernel<<<(tot + 255) / 256, blk, 0, stream>>>(kf, HKV, tot);
      }
      // values = swiglu(xn, wv1, wv2, wv3)
      gemm_bf16<false><<<dim3(PHID / 128, S / 128), blk, 0, stream>>>(
          xnbf, wv1 + (size_t)j * DIM * PHID, g1, nullptr, S, PHID, DIM);
      gemm_bf16<false><<<dim3(PHID / 128, S / 128), blk, 0, stream>>>(
          xnbf, wv3 + (size_t)j * DIM * PHID, g3, nullptr, S, PHID, DIM);
      silu_mul<<<(S * PHID + 255) / 256, blk, 0, stream>>>(g1, g3, actbf, S * PHID);
      gemm_bf16<false><<<dim3((HKV * HD) / 128, S / 128), blk, 0, stream>>>(
          actbf, wv2 + (size_t)j * PHID * HKV * HD, vf, nullptr, S, HKV * HD, PHID);
    }

    // attention (causal flash, WMMA), writes bf16 (S x 1024)
    flash_attn<<<dim3(S / 128, H), blk, 0, stream>>>(qf, kf, vf, aobf);

    // h = h + o @ wo[i]   (residual fused into GEMM epilogue)
    gemm_bf16<true><<<dim3(DIM / 128, S / 128), blk, 0, stream>>>(
        aobf, wo + (size_t)i * H * HD * DIM, hs, hs, S, DIM, H * HD);

    // FFN: h = h + swiglu(rmsnorm(h), w1, w2, w3)
    rmsnorm_bf16<<<S, blk, 0, stream>>>(hs, ffn_norm_w + (size_t)i * DIM, xnbf, DIM);
    gemm_bf16<false><<<dim3(FFN / 128, S / 128), blk, 0, stream>>>(
        xnbf, w1 + (size_t)i * DIM * FFN, g1, nullptr, S, FFN, DIM);
    gemm_bf16<false><<<dim3(FFN / 128, S / 128), blk, 0, stream>>>(
        xnbf, w3 + (size_t)i * DIM * FFN, g3, nullptr, S, FFN, DIM);
    silu_mul<<<(S * FFN + 255) / 256, blk, 0, stream>>>(g1, g3, actbf, S * FFN);
    gemm_bf16<true><<<dim3(DIM / 128, S / 128), blk, 0, stream>>>(
        actbf, w2 + (size_t)i * FFN * DIM, hs, hs, S, DIM, FFN);
  }

  // logits = rmsnorm(h, final_norm_w) @ out_w
  rmsnorm_bf16<<<S, blk, 0, stream>>>(hs, final_norm_w, xnbf, DIM);
  gemm_bf16<false><<<dim3(VOCAB / 128, S / 128), blk, 0, stream>>>(
      xnbf, out_w, logits, nullptr, S, VOCAB, DIM);
}
